// DiT_18056042512615
// MI455X (gfx1250) — matrix operand order repaired and checked by reading, not compiled, not run on hardware
//
#include <hip/hip_runtime.h>

// ---------------- common types / helpers ----------------
typedef __bf16 bf16_t;
typedef bf16_t v16bf __attribute__((ext_vector_type(16)));
typedef float  v8f   __attribute__((ext_vector_type(8)));
typedef unsigned int v8u __attribute__((ext_vector_type(8)));
typedef unsigned short u16_t;

#define DEV __device__ __forceinline__

constexpr int ZB = 64;   // batches
constexpr int D  = 128;  // model dim
constexpr int H  = 4;    // heads
constexpr int DK = 32;   // head dim
constexpr int NN = 512;  // tokens (8*8*8)

DEV u16_t f2bfu(float f) {                 // f32 -> bf16 bits via HW cvt (RNE)
  bf16_t h = (bf16_t)f;
  return __builtin_bit_cast(u16_t, h);
}
DEV v8f vzero() {
  v8f z = {0.f, 0.f, 0.f, 0.f, 0.f, 0.f, 0.f, 0.f};
  return z;
}
// B column gather: 16 bf16 down k (row stride), k in table order -> v16bf
DEV v16bf load_bf16x16(const u16_t* p, int stride) {
  int hk = (threadIdx.x & 16) ? 8 : 0;
  v16bf v;
#pragma unroll
  for (int i = 0; i < 8; ++i) {
    v[i]     = __builtin_bit_cast(bf16_t, p[(hk + i) * stride]);
    v[i + 8] = __builtin_bit_cast(bf16_t, p[(16 + hk + i) * stride]);
  }
  return v;
}
// two 8-halfs segments (A-matrix layout) -> v16bf
DEV v16bf load_bf16x8x2(const u16_t* p0, const u16_t* p1) {
  const unsigned int* q0 = (const unsigned int*)p0;
  const unsigned int* q1 = (const unsigned int*)p1;
  v8u u;
#pragma unroll
  for (int i = 0; i < 4; ++i) { u[i] = q0[i]; u[i + 4] = q1[i]; }
  return __builtin_bit_cast(v16bf, u);
}
DEV v8f wmma_bf16(v16bf a, v16bf b, v8f c) {
  return __builtin_amdgcn_wmma_f32_16x16x32_bf16(false, a, false, b, (short)0, c,
                                                 false, false);
}
template <int MASK>
DEV float swz_xor(float v) {  // ds_swizzle group-of-32 xor (and_mask=0x1F)
  int r = __builtin_amdgcn_ds_swizzle(__builtin_bit_cast(int, v),
                                      (MASK << 10) | 0x1F);
  return __builtin_bit_cast(float, r);
}
DEV float redmax16(float v) {
  v = fmaxf(v, swz_xor<1>(v)); v = fmaxf(v, swz_xor<2>(v));
  v = fmaxf(v, swz_xor<4>(v)); v = fmaxf(v, swz_xor<8>(v));
  return v;
}
DEV float redsum16(float v) {
  v += swz_xor<1>(v); v += swz_xor<2>(v);
  v += swz_xor<4>(v); v += swz_xor<8>(v);
  return v;
}
// async global->LDS copy of one dword (CDNA5 GLOBAL_LOAD_ASYNC_TO_LDS_B32)
DEV void async_g2l_b32(const u16_t* gsrc, const u16_t* ldst) {
  unsigned      lds_off = (unsigned)(size_t)ldst;  // generic LDS ptr low 32b
  unsigned long long g = (unsigned long long)(size_t)gsrc;
  asm volatile("global_load_async_to_lds_b32 %0, %1, off"
               :: "v"(lds_off), "v"(g) : "memory");
}
DEV void wait_async0() {
  asm volatile("s_wait_asynccnt 0x0" ::: "memory");
}

// ---------------- kernel 0: weight pre-conversion to bf16 ----------------
__global__ void k_prep(const float* __restrict__ qw, const float* __restrict__ kw,
                       const float* __restrict__ vw, const float* __restrict__ ow,
                       u16_t* __restrict__ qwb, u16_t* __restrict__ kwb,
                       u16_t* __restrict__ vwb, u16_t* __restrict__ owb) {
  int i = blockIdx.x * 256 + threadIdx.x;  // 0..16383
  const float qscale = 0.17677669529663687f;  // 1/sqrt(32) folded into Q weights
  qwb[i] = f2bfu(qw[i] * qscale);
  kwb[i] = f2bfu(kw[i]);
  vwb[i] = f2bfu(vw[i]);
  owb[i] = f2bfu(ow[i]);
}

// ---------------- kernel 1: cond MLPs ----------------
struct MlpPtrs { const float* p[24]; };

__global__ void k_cond(const float* __restrict__ nodes,
                       const float* __restrict__ tvec, MlpPtrs W,
                       float* __restrict__ condv) {
  int b = blockIdx.x, t = threadIdx.x;  // 128 threads
  __shared__ float vin[128], h1[128], h2[128];
  vin[t] = nodes[b * 128 + t] + tvec[b];
  __syncthreads();
  float* base = condv + (size_t)b * 6 * 128;
  for (int m = 0; m < 4; ++m) {
    const float* w1 = W.p[m * 6 + 0]; const float* b1 = W.p[m * 6 + 1];
    const float* w2 = W.p[m * 6 + 2]; const float* b2 = W.p[m * 6 + 3];
    const float* w3 = W.p[m * 6 + 4]; const float* b3 = W.p[m * 6 + 5];
    float a = b1[t];
    for (int i = 0; i < 128; ++i) a += vin[i] * w1[i * 128 + t];
    a = a / (1.f + __expf(-a));
    h1[t] = a; __syncthreads();
    a = b2[t];
    for (int i = 0; i < 128; ++i) a += h1[i] * w2[i * 128 + t];
    a = a / (1.f + __expf(-a));
    h2[t] = a; __syncthreads();
    bool gb = (m == 0) || (m == 2);
    int dout = gb ? 256 : 128;
    float o0 = b3[t];
    float o1 = gb ? b3[t + 128] : 0.f;
    for (int i = 0; i < 128; ++i) {
      float hv = h2[i];
      o0 += hv * w3[i * dout + t];
      if (gb) o1 += hv * w3[i * dout + t + 128];
    }
    // bug-repro unpack: alpha = first half of gb, gamma = second half, beta = a-MLP
    if (m == 0)      { base[0 * 128 + t] = o0; base[1 * 128 + t] = o1; } // alpha1,gamma1
    else if (m == 1) { base[2 * 128 + t] = o0; }                        // beta1
    else if (m == 2) { base[3 * 128 + t] = o0; base[4 * 128 + t] = o1; }// alpha2,gamma2
    else             { base[5 * 128 + t] = o0; }                        // beta2
    __syncthreads();
  }
}

// ---------------- kernel 2: snorm + modulate -> bf16 x2 ----------------
__global__ void k_norm1(const float* __restrict__ latent,
                        const float* __restrict__ condv,
                        u16_t* __restrict__ x2) {
  int b = blockIdx.x, rb = blockIdx.y * 64, t = threadIdx.x;  // 64 threads
  __shared__ float tile[64 * 129];
  __shared__ float mean_s[64], inv_s[64];
  const float* lb = latent + (size_t)b * D * NN;
  float s = 0.f, s2 = 0.f;
  for (int d = 0; d < 128; ++d) {       // coalesced: lanes span n
    float v = lb[(size_t)d * NN + rb + t];
    tile[t * 129 + d] = v; s += v; s2 += v * v;
  }
  float mean = s * (1.f / 128.f);
  float var = (s2 - 128.f * mean * mean) * (1.f / 127.f);
  float sd = sqrtf(fmaxf(var, 0.f));
  if (sd == 0.f) sd = 1.f;
  mean_s[t] = mean; inv_s[t] = 1.f / sd;
  __syncthreads();
  const float* gamma = condv + (size_t)b * 6 * 128 + 1 * 128;
  const float* beta  = condv + (size_t)b * 6 * 128 + 2 * 128;
  float g0 = gamma[t], g1 = gamma[t + 64], be0 = beta[t], be1 = beta[t + 64];
  for (int r = 0; r < 64; ++r) {        // coalesced: lanes span d
    float m = mean_s[r], iv = inv_s[r];
    float v0 = (tile[r * 129 + t]      - m) * iv * g0 + be0;
    float v1 = (tile[r * 129 + t + 64] - m) * iv * g1 + be1;
    size_t o = (size_t)(b * NN + rb + r) * 128;
    x2[o + t]      = f2bfu(v0);
    x2[o + t + 64] = f2bfu(v1);
  }
}

// ---------------- kernel 3: QKV projection (WMMA, bf16 weights) ----------
__global__ void k_qkv(const u16_t* __restrict__ x2, const u16_t* __restrict__ qwb,
                      const u16_t* __restrict__ kwb, const u16_t* __restrict__ vwb,
                      const float* __restrict__ qb, const float* __restrict__ kb,
                      const float* __restrict__ vb, u16_t* __restrict__ Qo,
                      u16_t* __restrict__ Kt, u16_t* __restrict__ Vo) {
  int blk = blockIdx.x;
  int mt = blk & 3, h = (blk >> 2) & 3, b = blk >> 4;
  int w = threadIdx.x >> 5, lane = threadIdx.x & 31;
  int rbase = mt * 128 + w * 16;
  int half8 = (lane < 16) ? 0 : 8;
  const u16_t* xrow = x2 + (size_t)(b * NN + rbase + (lane & 15)) * 128;
  const float qscale = 0.17677669529663687f;
  v8f cq0 = vzero(), cq1 = vzero(), ck0 = vzero(), ck1 = vzero(),
      cv0 = vzero(), cv1 = vzero();
#pragma unroll
  for (int ks = 0; ks < 4; ++ks) {
    v16bf a = load_bf16x8x2(xrow + ks * 32 + half8, xrow + ks * 32 + half8 + 16);
    size_t wb = ((size_t)h * 128 + ks * 32) * 32 + (lane & 15);  // B: lane = N column
    cq0 = wmma_bf16(a, load_bf16x16(qwb + wb, 32), cq0);
    cq1 = wmma_bf16(a, load_bf16x16(qwb + wb + 16, 32), cq1);
    ck0 = wmma_bf16(a, load_bf16x16(kwb + wb, 32), ck0);
    ck1 = wmma_bf16(a, load_bf16x16(kwb + wb + 16, 32), ck1);
    cv0 = wmma_bf16(a, load_bf16x16(vwb + wb, 32), cv0);
    cv1 = wmma_bf16(a, load_bf16x16(vwb + wb + 16, 32), cv1);
  }
  size_t hb = ((size_t)(b * H + h)) * NN * DK;
#pragma unroll
  for (int r = 0; r < 8; ++r) {
    int m = rbase + r + ((lane < 16) ? 0 : 8);
    int c0 = (lane & 15), c1 = 16 + (lane & 15);
    float q0 = cq0[r] + qb[h * 32 + c0] * qscale;
    float q1 = cq1[r] + qb[h * 32 + c1] * qscale;
    float k0 = ck0[r] + kb[h * 32 + c0];
    float k1 = ck1[r] + kb[h * 32 + c1];
    float v0 = cv0[r] + vb[h * 32 + c0];
    float v1 = cv1[r] + vb[h * 32 + c1];
    Qo[hb + (size_t)m * 32 + c0] = f2bfu(q0);
    Qo[hb + (size_t)m * 32 + c1] = f2bfu(q1);
    Kt[hb + (size_t)c0 * NN + m] = f2bfu(k0);  // K transposed [dk][n]
    Kt[hb + (size_t)c1 * NN + m] = f2bfu(k1);
    Vo[hb + (size_t)m * 32 + c0] = f2bfu(v0);
    Vo[hb + (size_t)m * 32 + c1] = f2bfu(v1);
  }
}

// ------- kernel 4: flash attention (WMMA + async LDS copy + ds_swizzle) -----
__global__ void k_attn(const u16_t* __restrict__ Qo, const u16_t* __restrict__ Kt,
                       const u16_t* __restrict__ Vo, u16_t* __restrict__ Oo) {
  int blk = blockIdx.x;
  int mt = blk & 3, h = (blk >> 2) & 3, b = blk >> 4;
  int w = threadIdx.x >> 5, lane = threadIdx.x & 31;
  __shared__ __align__(16) u16_t skt[32 * 130];    // Kt chunk [32][128]+pad
  __shared__ __align__(16) u16_t sv[128 * 34];     // V  chunk [128][32]+pad
  __shared__ __align__(16) u16_t sp[8 * 16 * 34];  // per-wave P buffer
  size_t hb = ((size_t)(b * H + h)) * NN * DK;
  int rbase = mt * 128 + w * 16;
  int half8 = (lane < 16) ? 0 : 8;
  const u16_t* qrow = Qo + hb + (size_t)(rbase + (lane & 15)) * 32;
  v16bf qa = load_bf16x8x2(qrow + half8, qrow + half8 + 16);
  v8f o0 = vzero(), o1 = vzero();
  float mrun[8], lrun[8];
#pragma unroll
  for (int r = 0; r < 8; ++r) { mrun[r] = -3.0e38f; lrun[r] = 0.f; }
  u16_t* pw = sp + w * 16 * 34;
  for (int ch = 0; ch < 4; ++ch) {
    // async DMA chunk staging: Kt[32][128c..], V[128c..][32]  (b32 per lane)
#pragma unroll
    for (int i = 0; i < 8; ++i) {
      int e = (threadIdx.x + 256 * i) * 2;  // even u16 index, 2 elems per b32
      int rr = e >> 7, cc = e & 127;
      async_g2l_b32(Kt + hb + (size_t)rr * NN + ch * 128 + cc,
                    skt + rr * 130 + cc);
      int rv = e >> 5, cv2 = e & 31;
      async_g2l_b32(Vo + hb + (size_t)(ch * 128 + rv) * 32 + cv2,
                    sv + rv * 34 + cv2);
    }
    wait_async0();
    __syncthreads();
    for (int pr = 0; pr < 4; ++pr) {  // 32 columns per step
      const u16_t* kt0 = skt + pr * 32 + (lane & 15);
      v16bf bk0 = load_bf16x16(kt0, 130);
      v16bf bk1 = load_bf16x16(kt0 + 16, 130);
      v8f s0 = wmma_bf16(qa, bk0, vzero());
      v8f s1 = wmma_bf16(qa, bk1, vzero());
      float corr[8];
#pragma unroll
      for (int r = 0; r < 8; ++r) {  // online softmax per row
        float tmax = redmax16(fmaxf(s0[r], s1[r]));
        float mnew = fmaxf(mrun[r], tmax);
        float p0 = __expf(s0[r] - mnew);
        float p1 = __expf(s1[r] - mnew);
        float cr = __expf(mrun[r] - mnew);
        float rs = redsum16(p0 + p1);
        lrun[r] = lrun[r] * cr + rs;
        mrun[r] = mnew;
        corr[r] = cr;
        s0[r] = p0; s1[r] = p1;
      }
#pragma unroll
      for (int r = 0; r < 8; ++r) { o0[r] *= corr[r]; o1[r] *= corr[r]; }
      // P tile -> wave-private LDS (C layout -> row major)
#pragma unroll
      for (int r = 0; r < 8; ++r) {
        int row = r + ((lane < 16) ? 0 : 8);
        pw[row * 34 + (lane & 15)]      = f2bfu(s0[r]);
        pw[row * 34 + 16 + (lane & 15)] = f2bfu(s1[r]);
      }
      const u16_t* prow = pw + (lane & 15) * 34;  // reload in A layout
      v16bf pa = load_bf16x8x2(prow + half8, prow + half8 + 16);
      const u16_t* vrow = sv + (pr * 32) * 34 + (lane & 15);
      v16bf bv0 = load_bf16x16(vrow, 34);
      v16bf bv1 = load_bf16x16(vrow + 16, 34);
      o0 = wmma_bf16(pa, bv0, o0);
      o1 = wmma_bf16(pa, bv1, o1);
    }
    __syncthreads();
  }
#pragma unroll
  for (int r = 0; r < 8; ++r) {
    int n = rbase + r + ((lane < 16) ? 0 : 8);
    float inv = 1.f / lrun[r];
    int k0 = (lane & 15), k1 = 16 + (lane & 15);
    // reference permute(0,2,3,1): d = k*H + h
    Oo[(size_t)(b * NN + n) * 128 + k0 * 4 + h] = f2bfu(o0[r] * inv);
    Oo[(size_t)(b * NN + n) * 128 + k1 * 4 + h] = f2bfu(o1[r] * inv);
  }
}

// ---------------- kernel 5: out-projection + residual (WMMA) ----------------
__global__ void k_proj(const u16_t* __restrict__ Oo, const u16_t* __restrict__ owb,
                       const float* __restrict__ latent,
                       const float* __restrict__ condv, float* __restrict__ xr) {
  int blk = blockIdx.x;
  int mt = blk & 3, b = blk >> 2;
  int w = threadIdx.x >> 5, lane = threadIdx.x & 31;
  int rbase = mt * 128 + w * 16;
  int half8 = (lane < 16) ? 0 : 8;
  const u16_t* orow = Oo + (size_t)(b * NN + rbase + (lane & 15)) * 128;
  v8f acc[8];
#pragma unroll
  for (int i = 0; i < 8; ++i) acc[i] = vzero();
#pragma unroll
  for (int ks = 0; ks < 4; ++ks) {
    v16bf a = load_bf16x8x2(orow + ks * 32 + half8, orow + ks * 32 + half8 + 16);
    const u16_t* wrow = owb + (size_t)(ks * 32) * 128 + (lane & 15);
#pragma unroll
    for (int nt = 0; nt < 8; ++nt)
      acc[nt] = wmma_bf16(a, load_bf16x16(wrow + nt * 16, 128), acc[nt]);
  }
  const float* alpha1 = condv + (size_t)b * 6 * 128 + 0 * 128;
  const float* lb = latent + (size_t)b * D * NN;
#pragma unroll
  for (int nt = 0; nt < 8; ++nt) {
    int d = nt * 16 + (lane & 15);
    float al = alpha1[d];
#pragma unroll
    for (int r = 0; r < 8; ++r) {
      int n = rbase + r + ((lane < 16) ? 0 : 8);
      float xv = lb[(size_t)d * NN + n];
      xr[(size_t)(b * NN + n) * 128 + d] = xv + al * acc[nt][r];
    }
  }
}

// ---------------- kernel 6: second snorm + residual + transpose out ---------
__global__ void k_final(const float* __restrict__ xr,
                        const float* __restrict__ condv,
                        float* __restrict__ outp) {
  int b = blockIdx.x, rb = blockIdx.y * 64, t = threadIdx.x;  // 64 threads
  __shared__ float tile[64 * 129];
  const float* xrb = xr + (size_t)(b * NN + rb) * 128;
  for (int r = 0; r < 64; ++r) {        // coalesced: lanes span d
    tile[r * 129 + t]      = xrb[(size_t)r * 128 + t];
    tile[r * 129 + t + 64] = xrb[(size_t)r * 128 + t + 64];
  }
  __syncthreads();
  float s = 0.f, s2 = 0.f;
  for (int d = 0; d < 128; ++d) { float v = tile[t * 129 + d]; s += v; s2 += v * v; }
  float mean = s * (1.f / 128.f);
  float var = (s2 - 128.f * mean * mean) * (1.f / 127.f);
  float sd = sqrtf(fmaxf(var, 0.f));
  if (sd == 0.f) sd = 1.f;
  float iv = 1.f / sd;
  const float* base = condv + (size_t)b * 6 * 128;
  float* ob = outp + (size_t)b * D * NN;
  for (int d = 0; d < 128; ++d) {       // coalesced: lanes span n
    float al = base[3 * 128 + d], g = base[4 * 128 + d], be = base[5 * 128 + d];
    float v = tile[t * 129 + d];
    float nv = (v - mean) * iv * g + be;
    ob[(size_t)d * NN + rb + t] = v + al * nv;
  }
}

// ---------------- host launcher ----------------
extern "C" void kernel_launch(void* const* d_in, const int* in_sizes, int n_in,
                              void* d_out, int out_size, void* d_ws,
                              size_t ws_size, hipStream_t stream) {
  const float* latent = (const float*)d_in[0];
  const float* nodes  = (const float*)d_in[1];
  const float* tv     = (const float*)d_in[2];
  const float* qw     = (const float*)d_in[3];
  const float* kw     = (const float*)d_in[4];
  const float* vw     = (const float*)d_in[5];
  const float* qb     = (const float*)d_in[6];
  const float* kb     = (const float*)d_in[7];
  const float* vb     = (const float*)d_in[8];
  const float* ow     = (const float*)d_in[9];
  MlpPtrs W;
  for (int i = 0; i < 24; ++i) W.p[i] = (const float*)d_in[10 + i];

  char* w = (char*)d_ws;
  float* condv = (float*)w;  w += (size_t)ZB * 6 * 128 * 4;        // 192 KB
  u16_t* x2 = (u16_t*)w;     w += (size_t)ZB * NN * D * 2;         // 8 MB
  u16_t* Qo = (u16_t*)w;     w += (size_t)ZB * H * NN * DK * 2;    // 8 MB
  u16_t* Kt = (u16_t*)w;     w += (size_t)ZB * H * NN * DK * 2;    // 8 MB
  u16_t* Vo = (u16_t*)w;     w += (size_t)ZB * H * NN * DK * 2;    // 8 MB
  u16_t* Oo = (u16_t*)w;     w += (size_t)ZB * NN * D * 2;         // 8 MB
  float* xr = (float*)w;     w += (size_t)ZB * NN * D * 4;         // 16 MB
  u16_t* qwb = (u16_t*)w;    w += (size_t)H * D * DK * 2;          // 32 KB
  u16_t* kwb = (u16_t*)w;    w += (size_t)H * D * DK * 2;
  u16_t* vwb = (u16_t*)w;    w += (size_t)H * D * DK * 2;
  u16_t* owb = (u16_t*)w;    w += (size_t)D * D * 2;

  k_prep<<<64, 256, 0, stream>>>(qw, kw, vw, ow, qwb, kwb, vwb, owb);
  k_cond<<<ZB, 128, 0, stream>>>(nodes, tv, W, condv);
  k_norm1<<<dim3(ZB, 8), 64, 0, stream>>>(latent, condv, x2);
  k_qkv<<<ZB * H * 4, 256, 0, stream>>>(x2, qwb, kwb, vwb, qb, kb, vb, Qo, Kt, Vo);
  k_attn<<<ZB * H * 4, 256, 0, stream>>>(Qo, Kt, Vo, Oo);
  k_proj<<<ZB * 4, 256, 0, stream>>>(Oo, owb, latent, condv, xr);
  k_final<<<dim3(ZB, 8), 64, 0, stream>>>(xr, condv, (float*)d_out);
}
